// MoEBlock_78606491451538
// MI455X (gfx1250) — compile-verified, matrix operand order checked by reading, not run on hardware
//
#include <hip/hip_runtime.h>
#include <hip/hip_bf16.h>

// ---------------------------------------------------------------------------
// MoE block for MI455X (gfx1250): bf16 WMMA expert FFN, deterministic routing.
// B=4 S=2048 H=1024 F=4096 E=8 K=2  -> T=8192, capacity=2048
// MT=32 token tiles (halves L2 weight traffic vs MT=16), async-LDS x staging.
// ---------------------------------------------------------------------------

#define T_TOK   8192
#define H_DIM   1024
#define F_DIM   4096
#define E_NUM   8
#define K_TOP   2
#define CAPACITY 2048
#define MT      32            // token tile (M) per workgroup (2 WMMA M-subtiles)
#define FHALF   2048          // F processed in two halves (LDS budget)
#define AUX_COEF 0.01f

typedef __bf16 bf16_t;
typedef __bf16 v16bf __attribute__((ext_vector_type(16)));
typedef float  v8f   __attribute__((ext_vector_type(8)));

// ---------------------------------------------------------------------------
// Init: zero output accumulator + routing accumulators (ws is not re-poisoned
// between replays, so we must re-zero every call).
// ---------------------------------------------------------------------------
__global__ __launch_bounds__(256) void moe_init_kernel(
    float* __restrict__ out, int* __restrict__ counts, float* __restrict__ msum,
    int* __restrict__ list_count)
{
    int idx = blockIdx.x * blockDim.x + threadIdx.x;
    for (int i = idx; i < T_TOK * H_DIM; i += gridDim.x * blockDim.x)
        out[i] = 0.0f;
    if (idx < E_NUM) {
        counts[idx] = 0;
        msum[idx] = 0.0f;
        list_count[idx] = 0;
    }
}

// ---------------------------------------------------------------------------
// Gating: one wave32 per token. Wg (1024x8 = 32KB) staged in LDS.
// ---------------------------------------------------------------------------
__global__ __launch_bounds__(256) void moe_gate_kernel(
    const float* __restrict__ x, const float* __restrict__ Wg,
    const float* __restrict__ bg,
    int* __restrict__ e0a, int* __restrict__ e1a,
    float* __restrict__ s0a, float* __restrict__ s1a,
    int* __restrict__ counts, float* __restrict__ msum)
{
    __shared__ float sWg[H_DIM * E_NUM];   // 32 KB
    int tid = threadIdx.x;
    for (int i = tid; i < H_DIM * E_NUM; i += 256) sWg[i] = Wg[i];
    __syncthreads();

    int wave = tid >> 5;
    int lane = tid & 31;
    int t = blockIdx.x * 8 + wave;          // 8 tokens per block

    float acc[E_NUM];
#pragma unroll
    for (int e = 0; e < E_NUM; ++e) acc[e] = 0.0f;

    const float* xp = x + (size_t)t * H_DIM;
    for (int k = lane; k < H_DIM; k += 32) {
        float xv = xp[k];
#pragma unroll
        for (int e = 0; e < E_NUM; ++e)
            acc[e] += xv * sWg[k * E_NUM + e];
    }
#pragma unroll
    for (int e = 0; e < E_NUM; ++e) {
#pragma unroll
        for (int off = 16; off > 0; off >>= 1)
            acc[e] += __shfl_xor(acc[e], off, 32);
    }

    if (lane == 0) {
        float l[E_NUM];
#pragma unroll
        for (int e = 0; e < E_NUM; ++e) l[e] = acc[e] + bg[e];

        int i0 = 0;
#pragma unroll
        for (int e = 1; e < E_NUM; ++e) if (l[e] > l[i0]) i0 = e;
        int i1 = (i0 == 0) ? 1 : 0;
#pragma unroll
        for (int e = 0; e < E_NUM; ++e)
            if (e != i0 && l[e] > l[i1]) i1 = e;

        float d = __expf(l[i1] - l[i0]);
        float s0 = 1.0f / (1.0f + d);
        float s1 = 1.0f - s0;

        float lmax = l[i0];
        float psum = 0.0f;
        float p[E_NUM];
#pragma unroll
        for (int e = 0; e < E_NUM; ++e) { p[e] = __expf(l[e] - lmax); psum += p[e]; }
        float inv = 1.0f / psum;
#pragma unroll
        for (int e = 0; e < E_NUM; ++e)
            atomicAdd(&msum[e], p[e] * inv);

        atomicAdd(&counts[i0], 1);
        atomicAdd(&counts[i1], 1);

        e0a[t] = i0; e1a[t] = i1;
        s0a[t] = s0; s1a[t] = s1;
    }
}

// ---------------------------------------------------------------------------
// Deterministic routing: one block per expert, block-wide prefix scan over the
// flattened (token, k) stream in reference order; truncate at CAPACITY.
// ---------------------------------------------------------------------------
__global__ __launch_bounds__(256) void moe_assign_kernel(
    const int* __restrict__ e0a, const int* __restrict__ e1a,
    const float* __restrict__ s0a, const float* __restrict__ s1a,
    int* __restrict__ list_tok, float* __restrict__ list_score,
    int* __restrict__ list_count)
{
    __shared__ int scan[256];
    int e = blockIdx.x;
    int tid = threadIdx.x;
    int base = 0;

    for (int start = 0; start < T_TOK * K_TOP; start += 256) {
        int idx = start + tid;
        int t = idx >> 1;
        int j = idx & 1;
        int ex = j ? e1a[t] : e0a[t];
        float sc = j ? s1a[t] : s0a[t];
        int flag = (ex == e) ? 1 : 0;

        scan[tid] = flag;
        __syncthreads();
        for (int off = 1; off < 256; off <<= 1) {
            int v = (tid >= off) ? scan[tid - off] : 0;
            __syncthreads();
            scan[tid] += v;
            __syncthreads();
        }
        int incl = scan[tid];
        int total = scan[255];
        if (flag) {
            int pos = base + incl - 1;
            if (pos < CAPACITY) {
                list_tok[e * CAPACITY + pos] = t;
                list_score[e * CAPACITY + pos] = sc;
            }
        }
        base += total;
        __syncthreads();
    }
    if (tid == 0) list_count[e] = (base < CAPACITY) ? base : CAPACITY;
}

// ---------------------------------------------------------------------------
// Expert FFN: one workgroup (8 wave32) per 32-token tile.
// Two passes over F halves:
//   stage 1: h = relu(x(32xH) @ W1(Hx2048) + b1)  -> sH (bf16, 128KB LDS)
//   stage 2: oacc += h(32x2048) @ W2(2048xH)      -> VGPR accumulators
// Final: o = oacc + b2, scale by score, atomic scatter-add.
// x tile staged via GLOBAL_LOAD_ASYNC_TO_LDS_B128 into a raw-fp32 region that
// aliases sH, then converted to bf16 (sX) before the GEMM passes.
// ---------------------------------------------------------------------------
__global__ __launch_bounds__(256) void moe_ffn_kernel(
    const float* __restrict__ x,
    const float* __restrict__ W1, const float* __restrict__ b1,
    const float* __restrict__ W2, const float* __restrict__ b2,
    const int* __restrict__ list_tok, const float* __restrict__ list_score,
    const int* __restrict__ list_count,
    float* __restrict__ out)
{
    extern __shared__ char smem[];
    bf16_t* sX = (bf16_t*)smem;                        // MT*H_DIM bf16  (64KB)
    bf16_t* sH = (bf16_t*)(smem + MT * H_DIM * 2);     // MT*FHALF bf16 (128KB)
    float*  sXraw = (float*)sH;                        // alias: MT*H_DIM f32 (128KB)
    int*   sTok = (int*)(smem + MT * H_DIM * 2 + MT * FHALF * 2);
    float* sSc  = (float*)(smem + MT * H_DIM * 2 + MT * FHALF * 2 + MT * 4);

    int e = blockIdx.y;
    int cnt = list_count[e];
    int row0 = blockIdx.x * MT;
    if (row0 >= cnt) return;

    int tid = threadIdx.x;
    int wave = tid >> 5;
    int lane = tid & 31;
    int half = lane >> 4;      // K half select (0: K 0..7,16..23; 1: K 8..15,24..31)
    int lr   = lane & 15;      // M (A rows) or N (B cols / C cols)

    if (tid < MT) {
        int r = row0 + tid;
        int tok = (r < cnt) ? list_tok[e * CAPACITY + r] : -1;
        sTok[tid] = tok;
        sSc[tid]  = (r < cnt) ? list_score[e * CAPACITY + r] : 0.0f;
    }
    __syncthreads();

    // ---- async stage: x tile (fp32) -> sXraw via GLOBAL_LOAD_ASYNC_TO_LDS ----
    {
        unsigned ldsbase = (unsigned)(size_t)sXraw;
#pragma unroll 1
        for (int it = 0; it < (MT * H_DIM / 4) / 256; ++it) {   // 32 iters
            int idx4 = it * 256 + tid;
            int i = idx4 * 4;                 // element index (float4 aligned)
            int m = i >> 10;
            int k = i & (H_DIM - 1);
            int tok = sTok[m];
            if (tok >= 0) {
                unsigned ldsoff = ldsbase + (unsigned)i * 4u;
                unsigned goff = ((unsigned)tok << 12) + ((unsigned)k << 2);
                asm volatile("global_load_async_to_lds_b128 %0, %1, %2"
                             :: "v"(ldsoff), "v"(goff), "s"(x) : "memory");
            }
        }
        asm volatile("s_wait_asynccnt 0x0" ::: "memory");
    }
    __syncthreads();

    // convert fp32 staging -> bf16 sX (zero-fill invalid rows)
    for (int i = tid; i < MT * H_DIM; i += 256) {
        int m = i >> 10;
        float v = (sTok[m] >= 0) ? sXraw[i] : 0.0f;
        sX[i] = (bf16_t)v;
    }
    __syncthreads();

    const float* W1p = W1 + (size_t)e * H_DIM * F_DIM;
    const float* W2p = W2 + (size_t)e * F_DIM * H_DIM;

    // stage-2 output accumulators: [m-subtile][n-tile], live across F halves
    v8f oacc[2][8];
#pragma unroll
    for (int mt = 0; mt < 2; ++mt)
#pragma unroll
        for (int nt = 0; nt < 8; ++nt)
            oacc[mt][nt] = (v8f){};

    for (int fh = 0; fh < 2; ++fh) {
        int fbase = fh * FHALF;

        // ---------------- stage 1: h-half = relu(x @ W1[:, half] + b1) --------
        for (int nt = 0; nt < 16; ++nt) {               // wave's N tiles in half
            int f0 = fbase + wave * 16 + nt * 128;
            v8f acc0 = (v8f){};
            v8f acc1 = (v8f){};
            for (int k0 = 0; k0 < H_DIM; k0 += 32) {
                v16bf a0, a1, b;
#pragma unroll
                for (int v = 0; v < 8; ++v) {
                    int kk = k0 + ((v < 4) ? (half * 8 + 2 * v)
                                           : (16 + half * 8 + 2 * (v - 4)));
                    a0[2 * v]     = sX[lr * H_DIM + kk];
                    a0[2 * v + 1] = sX[lr * H_DIM + kk + 1];
                    a1[2 * v]     = sX[(16 + lr) * H_DIM + kk];
                    a1[2 * v + 1] = sX[(16 + lr) * H_DIM + kk + 1];
                    b[2 * v]      = (bf16_t)W1p[(size_t)kk * F_DIM + f0 + lr];
                    b[2 * v + 1]  = (bf16_t)W1p[(size_t)(kk + 1) * F_DIM + f0 + lr];
                }
                if (k0 + 32 < H_DIM)
                    __builtin_prefetch(&W1p[(size_t)(k0 + 32) * F_DIM + f0 + lr], 0, 1);
                acc0 = __builtin_amdgcn_wmma_f32_16x16x32_bf16(
                    false, a0, false, b, (short)0, acc0, false, false);
                acc1 = __builtin_amdgcn_wmma_f32_16x16x32_bf16(
                    false, a1, false, b, (short)0, acc1, false, false);
            }
            float bias = b1[e * F_DIM + f0 + lr];
            int col = (f0 - fbase) + lr;
#pragma unroll
            for (int r = 0; r < 8; ++r) {
                int m0 = r + half * 8;
                float h0v = acc0[r] + bias;
                float h1v = acc1[r] + bias;
                h0v = h0v > 0.0f ? h0v : 0.0f;
                h1v = h1v > 0.0f ? h1v : 0.0f;
                sH[m0 * FHALF + col] = (bf16_t)h0v;
                sH[(16 + m0) * FHALF + col] = (bf16_t)h1v;
            }
        }
        __syncthreads();

        // ---------------- stage 2: oacc += h-half @ W2[half, :] ---------------
        for (int k0 = 0; k0 < FHALF; k0 += 32) {
            v16bf a0, a1;
#pragma unroll
            for (int v = 0; v < 8; ++v) {
                int kk = k0 + ((v < 4) ? (half * 8 + 2 * v)
                                       : (16 + half * 8 + 2 * (v - 4)));
                a0[2 * v]     = sH[lr * FHALF + kk];
                a0[2 * v + 1] = sH[lr * FHALF + kk + 1];
                a1[2 * v]     = sH[(16 + lr) * FHALF + kk];
                a1[2 * v + 1] = sH[(16 + lr) * FHALF + kk + 1];
            }
#pragma unroll
            for (int nt = 0; nt < 8; ++nt) {            // reuse A across 8 N-tiles
                int h0 = wave * 16 + nt * 128;
                v16bf b;
#pragma unroll
                for (int v = 0; v < 8; ++v) {
                    int kk = k0 + ((v < 4) ? (half * 8 + 2 * v)
                                           : (16 + half * 8 + 2 * (v - 4)));
                    b[2 * v]     = (bf16_t)W2p[(size_t)(fbase + kk) * H_DIM + h0 + lr];
                    b[2 * v + 1] = (bf16_t)W2p[(size_t)(fbase + kk + 1) * H_DIM + h0 + lr];
                }
                oacc[0][nt] = __builtin_amdgcn_wmma_f32_16x16x32_bf16(
                    false, a0, false, b, (short)0, oacc[0][nt], false, false);
                oacc[1][nt] = __builtin_amdgcn_wmma_f32_16x16x32_bf16(
                    false, a1, false, b, (short)0, oacc[1][nt], false, false);
            }
        }
        __syncthreads();   // before next F-half overwrites sH
    }

    // ---------------- epilogue: bias, score scale, scatter-add ----------------
#pragma unroll
    for (int nt = 0; nt < 8; ++nt) {
        int h0 = wave * 16 + nt * 128;
        float bias = b2[e * H_DIM + h0 + lr];
#pragma unroll
        for (int mt = 0; mt < 2; ++mt) {
#pragma unroll
            for (int r = 0; r < 8; ++r) {
                int m = mt * 16 + r + half * 8;
                int tok = sTok[m];
                if (tok >= 0) {
                    float o = (oacc[mt][nt][r] + bias) * sSc[m];
                    atomicAdd(&out[(size_t)tok * H_DIM + h0 + lr], o);
                }
            }
        }
    }
}

// ---------------------------------------------------------------------------
// Aux loss finalization: aux = 0.01 * sum(f_i * m_i) / E
// ---------------------------------------------------------------------------
__global__ void moe_aux_kernel(const int* __restrict__ counts,
                               const float* __restrict__ msum,
                               float* __restrict__ aux_out)
{
    if (threadIdx.x == 0 && blockIdx.x == 0) {
        float s = 0.0f;
#pragma unroll
        for (int e = 0; e < E_NUM; ++e) {
            float fi = (float)counts[e] / (float)(T_TOK * K_TOP);
            float mi = msum[e] / (float)T_TOK;
            s += fi * mi;
        }
        aux_out[0] = AUX_COEF * s / (float)E_NUM;
    }
}

// ---------------------------------------------------------------------------
extern "C" void kernel_launch(void* const* d_in, const int* in_sizes, int n_in,
                              void* d_out, int out_size, void* d_ws, size_t ws_size,
                              hipStream_t stream) {
    (void)in_sizes; (void)n_in; (void)out_size; (void)ws_size;

    const float* x  = (const float*)d_in[0];
    const float* Wg = (const float*)d_in[1];
    const float* bg = (const float*)d_in[2];
    const float* W1 = (const float*)d_in[3];
    const float* b1 = (const float*)d_in[4];
    const float* W2 = (const float*)d_in[5];
    const float* b2 = (const float*)d_in[6];
    float* out = (float*)d_out;                  // [T*H] output, then [1] aux

    char* w = (char*)d_ws;
    int*   g_e0 = (int*)w;        w += T_TOK * 4;
    int*   g_e1 = (int*)w;        w += T_TOK * 4;
    float* g_s0 = (float*)w;      w += T_TOK * 4;
    float* g_s1 = (float*)w;      w += T_TOK * 4;
    int*   counts = (int*)w;      w += E_NUM * 4;
    float* msum = (float*)w;      w += E_NUM * 4;
    int*   list_count = (int*)w;  w += E_NUM * 4;
    int*   list_tok = (int*)w;    w += E_NUM * CAPACITY * 4;
    float* list_score = (float*)w;

    moe_init_kernel<<<2048, 256, 0, stream>>>(out, counts, msum, list_count);

    moe_gate_kernel<<<T_TOK / 8, 256, 0, stream>>>(
        x, Wg, bg, g_e0, g_e1, g_s0, g_s1, counts, msum);

    moe_assign_kernel<<<E_NUM, 256, 0, stream>>>(
        g_e0, g_e1, g_s0, g_s1, list_tok, list_score, list_count);

    // dynamic LDS: sX bf16 (64KB) + sH bf16 / x-staging f32 (128KB) + metadata
    size_t smem = (size_t)MT * H_DIM * 2 + (size_t)MT * FHALF * 2 + MT * 8;
    dim3 grid(CAPACITY / MT, E_NUM);   // expert-major scheduling -> L2 weight reuse
    moe_ffn_kernel<<<grid, 256, smem, stream>>>(
        x, W1, b1, W2, b2, list_tok, list_score, list_count, out);

    moe_aux_kernel<<<1, 32, 0, stream>>>(counts, msum, out + (size_t)T_TOK * H_DIM);
}
